// attn_4329327035017
// MI455X (gfx1250) — matrix operand order repaired and checked by reading, not compiled, not run on hardware
//
#include <hip/hip_runtime.h>
#include <hip/hip_bf16.h>

// ---------------------------------------------------------------------------
// Causal MHA forward for MI455X (gfx1250, wave32, WMMA 16x16x32 f16 -> f32)
// B=4, T=2048, C=1024, H=16, D=64
// Async (GLOBAL_LOAD_ASYNC_TO_LDS) double-buffered staging feeding WMMAs.
// ---------------------------------------------------------------------------

typedef __attribute__((ext_vector_type(16))) _Float16 v16h;
typedef __attribute__((ext_vector_type(8)))  _Float16 v8h;
typedef __attribute__((ext_vector_type(8)))  float    v8f;

// builtin's pointee type is a GCC-style 16B int vector
typedef int v4i_g __attribute__((vector_size(16)));
typedef __attribute__((address_space(1))) v4i_g* global_v4i_p;
typedef __attribute__((address_space(3))) v4i_g* lds_v4i_p;

#define N_EMBD   1024
#define N_HEAD   16
#define HEAD_DIM 64
#define BATCH    4
#define SEQ      2048
#define M_ROWS   (BATCH * SEQ)      // 8192

#if defined(__AMDGCN__) && __has_builtin(__builtin_amdgcn_global_load_async_to_lds_b128) && __has_builtin(__builtin_amdgcn_s_wait_asynccnt)
#define USE_ASYNC_LDS 1
#else
#define USE_ASYNC_LDS 0
#endif

__device__ __forceinline__ v8f zero8() {
    v8f z;
#pragma unroll
    for (int i = 0; i < 8; ++i) z[i] = 0.0f;
    return z;
}

// 16B memory -> LDS copy. Async path: no VGPR data, tracked by ASYNCcnt.
__device__ __forceinline__ void cp16(const _Float16* g, _Float16* l) {
#if USE_ASYNC_LDS
    __builtin_amdgcn_global_load_async_to_lds_b128(
        (global_v4i_p)g, (lds_v4i_p)l, 0, 0);
#else
    *(v8h*)l = *(const v8h*)g;
#endif
}

__device__ __forceinline__ void wait_async() {
#if USE_ASYNC_LDS
    __builtin_amdgcn_s_wait_asynccnt(0);
#elif defined(__AMDGCN__)
    __asm__ volatile("" ::: "memory");
#endif
}

// A-fragment (16x32, f16) per ISA layout:
//   lanes 0-15 : row=lane,    K chunks {0..7} and {16..23}
//   lanes 16-31: row=lane-16, K chunks {8..15} and {24..31}
__device__ __forceinline__ v16h load_frag_a(const _Float16* base, int stride, int lane) {
    const int half = lane >> 4;
    const int row  = lane & 15;
    const _Float16* p = base + row * stride + half * 8;
    v16h a;
    *((v8h*)&a)     = *(const v8h*)p;
    *((v8h*)&a + 1) = *(const v8h*)(p + 16);
    return a;
}

// B-fragment (32x16, f16): column n=lane%16, same K order as A:
// lanes 0-15 -> K {0..7,16..23}, lanes 16-31 -> K {8..15,24..31}.  `base` is [N][K] row-major.
__device__ __forceinline__ v16h load_frag_b(const _Float16* base, int stride, int lane) {
    const int half = lane >> 4;
    const int col  = lane & 15;
    const _Float16* p = base + col * stride + half * 8;
    v16h b;
    *((v8h*)&b)     = *(const v8h*)p;
    *((v8h*)&b + 1) = *(const v8h*)(p + 16);
    return b;
}

// ---------------------------------------------------------------------------
// elementwise f32 -> f16 cast
__global__ void cast_f16_kernel(const float* __restrict__ in,
                                _Float16* __restrict__ out, int n) {
    int i = blockIdx.x * blockDim.x + threadIdx.x;
    if (i < n) out[i] = (_Float16)in[i];
}

// f32 [K][N] -> f16 [N][K] (transpose + cast)
__global__ void cast_transpose_f16_kernel(const float* __restrict__ in,
                                          _Float16* __restrict__ out,
                                          int K, int N) {
    int i = blockIdx.x * blockDim.x + threadIdx.x;
    if (i < K * N) {
        int n = i / K;
        int k = i - n * K;
        out[i] = (_Float16)in[(size_t)k * N + n];
    }
}

// ---------------------------------------------------------------------------
// Tiled WMMA GEMM:  C[M,N] = A[M,K](f16) * Bt[N,K](f16)^T + bias[N]
// Block 256 threads (8 waves, 4x2 wave grid), tile 128x128, BK=32,
// double-buffered async LDS staging.
// MODE 0: scatter into q (scaled 1/sqrt(D)) / k ([B,H,T,D]) / vT ([B,H,D,T])
// MODE 1: write f32 to out + bias
template <int MODE>
__global__ __launch_bounds__(256) void gemm_wmma_kernel(
    const _Float16* __restrict__ A,
    const _Float16* __restrict__ Bt,
    const float*    __restrict__ bias,
    int M, int N, int K,
    _Float16* __restrict__ qbuf,
    _Float16* __restrict__ kbuf,
    _Float16* __restrict__ vtbuf,
    float*    __restrict__ outf)
{
    // rows padded to 40 halfs (80B, 16B-aligned) to stagger LDS banks
    __shared__ _Float16 As[2][128 * 40];
    __shared__ _Float16 Bs[2][128 * 40];

    const int tid  = threadIdx.x;
    const int lane = tid & 31;
    const int wave = tid >> 5;
    const int wm   = wave & 3;   // 4 waves along M  -> 32 rows each
    const int wn   = wave >> 2;  // 2 waves along N  -> 64 cols each
    const int m0   = blockIdx.y * 128;
    const int n0   = blockIdx.x * 128;

    const int srow = tid >> 1;              // 0..127
    const int sc0  = (tid & 1) * 16;        // 0 or 16

    auto stage = [&](int buf, int k0) {
        const _Float16* ga = A  + (size_t)(m0 + srow) * K + k0 + sc0;
        cp16(ga,     &As[buf][srow * 40 + sc0]);
        cp16(ga + 8, &As[buf][srow * 40 + sc0 + 8]);
        const _Float16* gb = Bt + (size_t)(n0 + srow) * K + k0 + sc0;
        cp16(gb,     &Bs[buf][srow * 40 + sc0]);
        cp16(gb + 8, &Bs[buf][srow * 40 + sc0 + 8]);
    };

    v8f acc[2][4];
#pragma unroll
    for (int i = 0; i < 2; ++i)
#pragma unroll
        for (int j = 0; j < 4; ++j) acc[i][j] = zero8();

    stage(0, 0);                            // prologue
    int buf = 0;
    for (int k0 = 0; k0 < K; k0 += 32, buf ^= 1) {
        wait_async();                       // our async stage of `buf` landed
        __syncthreads();                    // everyone's stage of `buf` landed
        if (k0 + 32 < K) stage(buf ^ 1, k0 + 32);   // prefetch next tile

        v16h af[2], bf[4];
#pragma unroll
        for (int i = 0; i < 2; ++i)
            af[i] = load_frag_a(&As[buf][(wm * 32 + i * 16) * 40], 40, lane);
#pragma unroll
        for (int j = 0; j < 4; ++j)
            bf[j] = load_frag_b(&Bs[buf][(wn * 64 + j * 16) * 40], 40, lane);

#pragma unroll
        for (int i = 0; i < 2; ++i)
#pragma unroll
            for (int j = 0; j < 4; ++j)
                acc[i][j] = __builtin_amdgcn_wmma_f32_16x16x32_f16(
                    false, af[i], false, bf[j], (short)0, acc[i][j], false, false);
        // next iteration's barrier protects `buf` from being overwritten
    }

    // epilogue. C/D layout: VGPR r -> row r + 8*(lane/16), col = lane%16
    const int colbase = n0 + wn * 64;
    const int rowbase = m0 + wm * 32;
    const int lcol    = lane & 15;
    const int rowoff  = (lane >> 4) * 8;
#pragma unroll
    for (int i = 0; i < 2; ++i) {
#pragma unroll
        for (int j = 0; j < 4; ++j) {
            const int n  = colbase + j * 16 + lcol;
            const float bv = bias[n];
#pragma unroll
            for (int r = 0; r < 8; ++r) {
                const int m = rowbase + i * 16 + rowoff + r;
                const float v = acc[i][j][r] + bv;
                if (MODE == 0) {
                    const int which = n >> 10;          // 0=q 1=k 2=v
                    const int c  = n & 1023;
                    const int h  = c >> 6;
                    const int dd = c & 63;
                    const int b  = m >> 11;
                    const int t  = m & 2047;
                    const size_t bh = (size_t)(b * N_HEAD + h);
                    if (which == 0)
                        qbuf[(bh * SEQ + t) * HEAD_DIM + dd] = (_Float16)(v * 0.125f);
                    else if (which == 1)
                        kbuf[(bh * SEQ + t) * HEAD_DIM + dd] = (_Float16)v;
                    else
                        vtbuf[(bh * HEAD_DIM + dd) * SEQ + t] = (_Float16)v;
                } else {
                    outf[(size_t)m * N + n] = v;
                }
            }
        }
    }
}

// ---------------------------------------------------------------------------
// Flash attention: block = 128 query rows (8 waves x 16 rows), 1 (b,h) pair.
// Keys streamed in blocks of 64, double-buffered async LDS staging.
__global__ __launch_bounds__(256) void flash_attn_kernel(
    const _Float16* __restrict__ qh,   // [B,H,T,D], pre-scaled by 1/sqrt(D)
    const _Float16* __restrict__ kh,   // [B,H,T,D]
    const _Float16* __restrict__ vt,   // [B,H,D,T]
    _Float16* __restrict__ y)          // [B*T, C]
{
    __shared__ _Float16 Ks[2][64 * 64];       // [key][d]
    __shared__ _Float16 Vs[2][64 * 64];       // [d][key]
    __shared__ _Float16 Ps[8][16 * 64];       // per-wave P scratch [row][key]

    const int tid  = threadIdx.x;
    const int lane = tid & 31;
    const int wave = tid >> 5;
    const int bh   = blockIdx.y;              // 0..63  (b*16 + h)
    const int qb   = blockIdx.x * 128;
    const int q0   = qb + wave * 16;
    const size_t qk_base = (size_t)bh * SEQ * HEAD_DIM;
    const size_t vt_base = (size_t)bh * HEAD_DIM * SEQ;

    const int srow = tid >> 2;                // 0..63
    const int sc0  = (tid & 3) * 16;          // 0,16,32,48

    auto stage = [&](int buf, int j0) {
        const _Float16* gk = kh + qk_base + (size_t)(j0 + srow) * HEAD_DIM + sc0;
        cp16(gk,     &Ks[buf][srow * 64 + sc0]);
        cp16(gk + 8, &Ks[buf][srow * 64 + sc0 + 8]);
        const _Float16* gv = vt + vt_base + (size_t)srow * SEQ + j0 + sc0;
        cp16(gv,     &Vs[buf][srow * 64 + sc0]);
        cp16(gv + 8, &Vs[buf][srow * 64 + sc0 + 8]);
    };

    // Q A-fragments for the two 32-wide K chunks of D=64 (loaded once)
    v16h qf[2];
    {
        const int half = lane >> 4;
        const int row  = lane & 15;
        const _Float16* qrow = qh + qk_base + (size_t)(q0 + row) * HEAD_DIM;
#pragma unroll
        for (int kk = 0; kk < 2; ++kk) {
            const _Float16* p = qrow + kk * 32 + half * 8;
            v16h a;
            *((v8h*)&a)     = *(const v8h*)p;
            *((v8h*)&a + 1) = *(const v8h*)(p + 16);
            qf[kk] = a;
        }
    }

    v8f o[4];
#pragma unroll
    for (int j = 0; j < 4; ++j) o[j] = zero8();
    float mrow[8], lrow[8];
#pragma unroll
    for (int r = 0; r < 8; ++r) { mrow[r] = -1e30f; lrow[r] = 0.0f; }

    const int lcol   = lane & 15;
    const int rowoff = (lane >> 4) * 8;
    const int jend   = qb + 128;

    stage(0, 0);                              // prologue
    int buf = 0;
    for (int j0 = 0; j0 < jend; j0 += 64, buf ^= 1) {
        wait_async();
        __syncthreads();
        if (j0 + 64 < jend) stage(buf ^ 1, j0 + 64);

        // S = Q K^T : 4 key sub-tiles of 16, each 2 WMMAs over D
        v8f s[4];
#pragma unroll
        for (int st = 0; st < 4; ++st) {
            v8f a8 = zero8();
#pragma unroll
            for (int kk = 0; kk < 2; ++kk) {
                v16h b = load_frag_b(&Ks[buf][(st * 16) * 64 + kk * 32], 64, lane);
                a8 = __builtin_amdgcn_wmma_f32_16x16x32_f16(
                    false, qf[kk], false, b, (short)0, a8, false, false);
            }
            s[st] = a8;
        }

        // causal mask (arithmetic; EXEC stays all-ones for WMMA)
#pragma unroll
        for (int st = 0; st < 4; ++st) {
            const int j = j0 + st * 16 + lcol;
#pragma unroll
            for (int r = 0; r < 8; ++r) {
                const int m = q0 + rowoff + r;
                if (j > m) s[st][r] = -1e30f;
            }
        }

        // online softmax: row max / sum via width-16 xor shuffles
#pragma unroll
        for (int r = 0; r < 8; ++r) {
            float mx = fmaxf(fmaxf(s[0][r], s[1][r]), fmaxf(s[2][r], s[3][r]));
#pragma unroll
            for (int off = 8; off >= 1; off >>= 1)
                mx = fmaxf(mx, __shfl_xor(mx, off, 16));
            const float mnew  = fmaxf(mrow[r], mx);
            const float alpha = __expf(mrow[r] - mnew);
            mrow[r] = mnew;
            float psum = 0.0f;
#pragma unroll
            for (int st = 0; st < 4; ++st) {
                const float p = __expf(s[st][r] - mnew);
                s[st][r] = p;
                psum += p;
            }
#pragma unroll
            for (int off = 8; off >= 1; off >>= 1)
                psum += __shfl_xor(psum, off, 16);
            lrow[r] = lrow[r] * alpha + psum;
#pragma unroll
            for (int j = 0; j < 4; ++j) o[j][r] *= alpha;
        }

        // spill P (f16) to per-wave LDS scratch, row-major [16][64]
        _Float16* ps = &Ps[wave][0];
#pragma unroll
        for (int st = 0; st < 4; ++st)
#pragma unroll
            for (int r = 0; r < 8; ++r)
                ps[(rowoff + r) * 64 + st * 16 + lcol] = (_Float16)s[st][r];
        // per-wave region; LDS ops are in-order within a wave -> no barrier

        // O += P @ V
#pragma unroll
        for (int kk = 0; kk < 2; ++kk) {
            v16h a = load_frag_a(ps + kk * 32, 64, lane);
#pragma unroll
            for (int dt = 0; dt < 4; ++dt) {
                v16h b = load_frag_b(&Vs[buf][(dt * 16) * 64 + kk * 32], 64, lane);
                o[dt] = __builtin_amdgcn_wmma_f32_16x16x32_f16(
                    false, a, false, b, (short)0, o[dt], false, false);
            }
        }
        // next iteration's barrier protects Ks/Vs[buf] from refill
    }

    // normalize, write y[b*T+t][h*64+d] (f16)
    {
        const int b = bh >> 4;
        const int h = bh & 15;
#pragma unroll
        for (int r = 0; r < 8; ++r) {
            const float inv = 1.0f / lrow[r];
            const int t = q0 + rowoff + r;
            const size_t base = ((size_t)(b * SEQ + t)) * N_EMBD + h * HEAD_DIM;
#pragma unroll
            for (int dt = 0; dt < 4; ++dt)
                y[base + dt * 16 + lcol] = (_Float16)(o[dt][r] * inv);
        }
    }
}

// ---------------------------------------------------------------------------
extern "C" void kernel_launch(void* const* d_in, const int* in_sizes, int n_in,
                              void* d_out, int out_size, void* d_ws, size_t ws_size,
                              hipStream_t stream) {
    const float* x      = (const float*)d_in[0];  // [4,2048,1024]
    const float* W_attn = (const float*)d_in[1];  // [1024,3072]
    const float* b_attn = (const float*)d_in[2];  // [3072]
    const float* W_proj = (const float*)d_in[3];  // [1024,1024]
    const float* b_proj = (const float*)d_in[4];  // [1024]
    float* out = (float*)d_out;

    char* ws = (char*)d_ws;
    _Float16* xh  = (_Float16*)ws;  ws += (size_t)M_ROWS * N_EMBD * 2;       // 16 MiB
    _Float16* Wat = (_Float16*)ws;  ws += (size_t)3 * N_EMBD * N_EMBD * 2;   //  6 MiB [3072][1024]
    _Float16* Wpt = (_Float16*)ws;  ws += (size_t)N_EMBD * N_EMBD * 2;       //  2 MiB [1024][1024]
    _Float16* qh  = (_Float16*)ws;  ws += (size_t)M_ROWS * N_EMBD * 2;       // 16 MiB [B,H,T,D]
    _Float16* kh  = (_Float16*)ws;  ws += (size_t)M_ROWS * N_EMBD * 2;       // 16 MiB [B,H,T,D]
    _Float16* vth = (_Float16*)ws;  ws += (size_t)M_ROWS * N_EMBD * 2;       // 16 MiB [B,H,D,T]
    _Float16* yh  = (_Float16*)ws;  ws += (size_t)M_ROWS * N_EMBD * 2;       // 16 MiB [B*T,C]

    const int nx = M_ROWS * N_EMBD;                       // 8388608
    cast_f16_kernel<<<(nx + 255) / 256, 256, 0, stream>>>(x, xh, nx);
    const int nwa = N_EMBD * 3 * N_EMBD;                  // 3145728
    cast_transpose_f16_kernel<<<(nwa + 255) / 256, 256, 0, stream>>>(
        W_attn, Wat, N_EMBD, 3 * N_EMBD);
    const int nwp = N_EMBD * N_EMBD;                      // 1048576
    cast_transpose_f16_kernel<<<(nwp + 255) / 256, 256, 0, stream>>>(
        W_proj, Wpt, N_EMBD, N_EMBD);

    // qkv = x @ W_attn + b_attn  -> scatter q/k/vT
    dim3 g0(3 * N_EMBD / 128, M_ROWS / 128);  // (24, 64)
    gemm_wmma_kernel<0><<<g0, 256, 0, stream>>>(
        xh, Wat, b_attn, M_ROWS, 3 * N_EMBD, N_EMBD, qh, kh, vth, nullptr);

    // fused causal attention
    dim3 g1(SEQ / 128, BATCH * N_HEAD);       // (16, 64)
    flash_attn_kernel<<<g1, 256, 0, stream>>>(qh, kh, vth, yh);

    // out = y @ W_proj + b_proj
    dim3 g2(N_EMBD / 128, M_ROWS / 128);      // (8, 64)
    gemm_wmma_kernel<1><<<g2, 256, 0, stream>>>(
        yh, Wpt, b_proj, M_ROWS, N_EMBD, N_EMBD, nullptr, nullptr, nullptr, out);
}
